// TrainableActivation_22213570855664
// MI455X (gfx1250) — matrix operand order repaired and checked by reading, not compiled, not run on hardware
//
#include <hip/hip_runtime.h>
#include <cstdint>

typedef _Float16     v16h __attribute__((ext_vector_type(16)));
typedef float        v8f  __attribute__((ext_vector_type(8)));
typedef unsigned int v4u  __attribute__((ext_vector_type(4)));
typedef int          v8i  __attribute__((ext_vector_type(8)));
typedef int          v4i  __attribute__((ext_vector_type(4)));

#define NW    63
#define KLUT  4096
#define XMIN  (-1.5f)
#define XMAX  (1.5f)
#define HW    9216           // 96*96 spatial plane
#define NCH   64
#define NB    16
#define SPLIT 3
#define CHUNK (HW / SPLIT)   // 3072 elements per block
#define A2S   480.5f         // 1/(2*sigma^2), sigma = 2/62

#if defined(__has_builtin)
#if __has_builtin(__builtin_amdgcn_tensor_load_to_lds) && __has_builtin(__builtin_amdgcn_s_wait_tensorcnt)
#define USE_TDM 1
#endif
#endif

// ---------------------------------------------------------------------------
// Kernel 1: LUT = Phi[4096 x 63] * W^T[63 x 64] via v_wmma_f32_16x16x32_f16.
// One wave produces a 16x16 (kpoints x channels) tile of the LUT, accumulating
// over the knot (K) dimension in two chunks of 32 (63 knots, padded).
// A-matrix lane layout (16-bit A 16x32, ISA 7.12.2):
//   lane L holds row M=L&15; element i -> K = (i<8 ? i : i+8) + 8*(L>=16)
// B-matrix (16-bit B 32x16): lane L holds column N=L&15;
//   element i -> K = (i<8 ? i : i+8) + 8*(L>=16)
// C/D f32 16x16: lane L -> N=L&15; VGPR r -> M = r + 8*(L>=16)
// ---------------------------------------------------------------------------
__global__ __launch_bounds__(256) void rbf_lut_wmma_kernel(
    const float* __restrict__ w,   // [64][63]
    float* __restrict__ lut) {     // [64][KLUT]
  const int wid   = (blockIdx.x * blockDim.x + threadIdx.x) >> 5;  // 0..1023
  const int lane  = threadIdx.x & 31;
  const int ktile = wid >> 2;      // 0..255
  const int ctile = wid & 3;       // 0..3
  const int m     = lane & 15;
  const int hi    = lane >> 4;

  const float h  = (XMAX - XMIN) / (float)(KLUT - 1);
  const float xk = XMIN + (float)(ktile * 16 + m) * h;

  v8f acc = {};
#pragma unroll
  for (int jb = 0; jb < 64; jb += 32) {
    v16h A, B;
#pragma unroll
    for (int i = 0; i < 16; ++i) {
      // A: phi values for row m (this lane's k-point)
      const int ja = jb + (i < 8 ? i : i + 8) + hi * 8;
      float phi = 0.0f;
      if (ja < NW) {
        const float mu = -1.0f + (float)ja * (1.0f / 31.0f);
        const float d  = xk - mu;
        phi = __expf(-A2S * d * d);
      }
      A[i] = (_Float16)phi;
      // B: weight for column n (= m), knot jw
      const int jw = jb + (i < 8 ? i : i + 8) + hi * 8;
      const float wv = (jw < NW) ? w[(ctile * 16 + m) * NW + jw] : 0.0f;
      B[i] = (_Float16)wv;
    }
    acc = __builtin_amdgcn_wmma_f32_16x16x32_f16(
        /*neg_a=*/false, A, /*neg_b=*/false, B,
        /*c_mod=*/(short)0, acc, /*reuse_a=*/false, /*reuse_b=*/false);
  }

  const int c = ctile * 16 + m;    // output channel (N = lane&15)
#pragma unroll
  for (int r = 0; r < 8; ++r) {
    const int kk = ktile * 16 + r + hi * 8;
    lut[c * KLUT + kk] = acc[r];
  }
}

// ---------------------------------------------------------------------------
// Kernel 2: stream x, per-element linear interpolation from the per-channel
// LUT staged in LDS. LUT (1MB total) is L2-resident; LDS fill uses the Tensor
// Data Mover (1-D descriptor) when available.
// ---------------------------------------------------------------------------
__device__ __forceinline__ float lut_eval(float xv, const float* slut) {
  const float scale = (float)(KLUT - 1) / (XMAX - XMIN);
  float t = (xv - XMIN) * scale;
  t = fminf(fmaxf(t, 0.0f), (float)(KLUT - 1));
  int i0 = (int)t;
  i0 = i0 > (KLUT - 2) ? (KLUT - 2) : i0;
  const float fr = t - (float)i0;
  const float a  = slut[i0];
  const float b  = slut[i0 + 1];
  return fmaf(fr, b - a, a);
}

__global__ __launch_bounds__(256) void rbf_apply_kernel(
    const float* __restrict__ x,
    const float* __restrict__ lut,   // [64][KLUT]
    float* __restrict__ out) {
  __shared__ float slut[KLUT];

  const int plane = blockIdx.x / SPLIT;           // b*64 + c
  const int part  = blockIdx.x % SPLIT;
  const int c     = plane & (NCH - 1);
  const int base  = plane * HW + part * CHUNK;
  const float* g  = lut + (size_t)c * KLUT;

#ifdef USE_TDM
  if (threadIdx.x < 32) {  // wave 0 issues the TDM op (per-wave instruction)
    const unsigned lds_off = (unsigned)(uintptr_t)(&slut[0]);  // low 32 bits of flat LDS addr = LDS offset
    const unsigned long long ga = (unsigned long long)(uintptr_t)g;
    // D# group 0: count=1 | lds_addr | global_addr[56:0] | type=2
    v4u g0 = { 1u,
               lds_off,
               (unsigned)(ga & 0xFFFFFFFFu),
               (unsigned)((ga >> 32) & 0x01FFFFFFu) | (2u << 30) };
    // D# group 1: data_size=4B, tensor_dim0=KLUT, tensor_dim1=1,
    //             tile_dim0=KLUT, tensor_dim0_stride=KLUT
    v8i g1 = { (int)(2u << 16),                              // data_size=2 (4B)
               (int)(((unsigned)KLUT & 0xFFFFu) << 16),      // tensor_dim0[15:0]
               (int)(((unsigned)KLUT >> 16) | (1u << 16)),   // dim0[31:16] | dim1[15:0]
               (int)(((unsigned)KLUT & 0xFFFFu) << 16),      // tile_dim0
               0,                                            // tile_dim1/2 unused
               (int)KLUT,                                    // tensor_dim0_stride[31:0]
               0, 0 };
    v4i z4 = {0, 0, 0, 0};
#if __clang_major__ >= 23
    v8i z8 = {0, 0, 0, 0, 0, 0, 0, 0};
    __builtin_amdgcn_tensor_load_to_lds(g0, g1, z4, z4, z8, 0);
#else
    __builtin_amdgcn_tensor_load_to_lds(g0, g1, z4, z4, 0);
#endif
    __builtin_amdgcn_s_wait_tensorcnt((unsigned short)0);
  }
  __syncthreads();
#else
  for (int i = threadIdx.x; i < KLUT; i += 256) slut[i] = g[i];
  __syncthreads();
#endif

  const float4* __restrict__ xin = reinterpret_cast<const float4*>(x + base);
  float4* __restrict__       oup = reinterpret_cast<float4*>(out + base);
#pragma unroll
  for (int k = 0; k < CHUNK / (256 * 4); ++k) {   // 3 iterations of float4
    const int idx = k * 256 + threadIdx.x;
    const float4 v = xin[idx];
    float4 r;
    r.x = lut_eval(v.x, slut);
    r.y = lut_eval(v.y, slut);
    r.z = lut_eval(v.z, slut);
    r.w = lut_eval(v.w, slut);
    oup[idx] = r;
  }
}

// ---------------------------------------------------------------------------
// Fallback (only if workspace is too small): brute-force 63 exps/element.
// ---------------------------------------------------------------------------
__global__ __launch_bounds__(256) void rbf_brute_kernel(
    const float* __restrict__ x, const float* __restrict__ w,
    float* __restrict__ out, int total) {
  const int i = blockIdx.x * blockDim.x + threadIdx.x;
  if (i >= total) return;
  const int c = (i / HW) & (NCH - 1);
  const float xv = x[i];
  float acc = 0.0f;
#pragma unroll 9
  for (int j = 0; j < NW; ++j) {
    const float mu = -1.0f + (float)j * (1.0f / 31.0f);
    const float d  = xv - mu;
    acc = fmaf(w[c * NW + j], __expf(-A2S * d * d), acc);
  }
  out[i] = acc;
}

extern "C" void kernel_launch(void* const* d_in, const int* in_sizes, int n_in,
                              void* d_out, int out_size, void* d_ws, size_t ws_size,
                              hipStream_t stream) {
  const float* x = (const float*)d_in[0];
  const float* w = (const float*)d_in[1];
  float* out = (float*)d_out;
  const size_t lut_bytes = (size_t)NCH * KLUT * sizeof(float);

  if (ws_size >= lut_bytes) {
    float* lut = (float*)d_ws;
    // 1024 waves = (KLUT/16) k-tiles * 4 c-tiles; 8 waves/block -> 128 blocks
    rbf_lut_wmma_kernel<<<dim3(128), dim3(256), 0, stream>>>(w, lut);
    rbf_apply_kernel<<<dim3(NB * NCH * SPLIT), dim3(256), 0, stream>>>(x, lut, out);
  } else {
    const int total = in_sizes[0];
    rbf_brute_kernel<<<dim3((total + 255) / 256), dim3(256), 0, stream>>>(x, w, out, total);
  }
}
